// my_net_65335042507033
// MI455X (gfx1250) — matrix operand order repaired and checked by reading, not compiled, not run on hardware
//
#include <hip/hip_runtime.h>
#include <stdint.h>

#define LN_EPS 1e-5f

typedef __attribute__((ext_vector_type(16))) __bf16 v16bf;
typedef __attribute__((ext_vector_type(8)))  float  v8f;

union ABReg { v16bf v; uint4 q[2]; };

#define WMMA_BF16(A, B, C) \
  __builtin_amdgcn_wmma_f32_16x16x32_bf16(false, (A), false, (B), (short)0, (C), false, false)

static __device__ __forceinline__ unsigned short f32_to_bf16_bits(float f) {
  uint32_t u = __float_as_uint(f);
  uint32_t r = u + 0x7FFFu + ((u >> 16) & 1u);   // round-to-nearest-even
  return (unsigned short)(r >> 16);
}

// ---- WMMA operand loaders (bf16, 16x16x32) ------------------------------
// A (16x32, MxK): lane l<16 -> row l, k = hi*8 + 0..7 (vgpr0..3) and
//                 16 + hi*8 + 0..7 (vgpr4..7); hi = lane>>4.
static __device__ __forceinline__ v16bf load_a_tile(const unsigned short* base,
                                                    int stride, int kbase, int lane) {
  int r = lane & 15, hi = lane >> 4;
  const unsigned short* p = base + r * stride + kbase + hi * 8;
  ABReg a;
  a.q[0] = *(const uint4*)(p);
  a.q[1] = *(const uint4*)(p + 16);
  return a.v;
}

// B (32x16, KxN): lane l<16 -> column l, k = hi*8 + 0..7 and 16 + hi*8 + 0..7.
// Weight rows (PyTorch [N][K]) are exactly B columns -> contiguous loads.
static __device__ __forceinline__ v16bf load_b_tile(const unsigned short* base,
                                                    int stride, int kbase, int lane) {
  int c = lane & 15, hi = lane >> 4;
  const unsigned short* p = base + c * stride + kbase + hi * 8;
  ABReg b;
  b.q[0] = *(const uint4*)(p);
  b.q[1] = *(const uint4*)(p + 16);
  return b.v;
}

// ---- weight pad/convert: fp32 [rs][cs] -> bf16 [rd][cd], zero padded ----
__global__ void k_pad_bf16(unsigned short* __restrict__ dst,
                           const float* __restrict__ src,
                           int rs, int cs, int rd, int cd) {
  int idx = blockIdx.x * 256 + threadIdx.x;
  if (idx >= rd * cd) return;
  int r = idx / cd, c = idx - r * cd;
  unsigned short v = 0;
  if (r < rs && c < cs) v = f32_to_bf16_bits(src[r * cs + c]);
  dst[idx] = v;
}

// ---- embed: h = tanh(x@emb_w^T + b) + posenc; also bf16 padded copy -----
__global__ void k_embed(const float* __restrict__ x, const float* __restrict__ ew,
                        const float* __restrict__ eb, float* __restrict__ h,
                        unsigned short* __restrict__ hb) {
  int idx = blockIdx.x * 256 + threadIdx.x;       // over 32768*64
  if (idx >= 32768 * 64) return;
  int t = idx >> 6, d = idx & 63;
  if (d >= 36) { hb[t * 64 + d] = 0; return; }
  float acc = eb[d];
  for (int j = 0; j < 6; ++j) acc += x[t * 6 + j] * ew[d * 6 + j];
  acc = tanhf(acc);
  int s = t & 511;
  float div = __powf(10000.f, (float)(d + 1) * (1.f / 36.f));
  float ang = (float)s / div;
  float pe = (d & 1) ? __cosf(ang) : __sinf(ang);
  float v = acc + pe;
  h[t * 36 + d] = v;
  hb[t * 64 + d] = f32_to_bf16_bits(v);
}

// ---- QKV projection: [32768x64]bf16 @ [112(->128)x64]bf16 -> qkv f32 ----
__global__ void k_qkv(const unsigned short* __restrict__ hb,
                      const unsigned short* __restrict__ wq,
                      const float* __restrict__ bias, float* __restrict__ qkv) {
  int wave = threadIdx.x >> 5, lane = threadIdx.x & 31;
  int mtile = blockIdx.x * 8 + wave;
  int row0 = mtile * 16;
  v16bf a0 = load_a_tile(hb + (size_t)row0 * 64, 64, 0, lane);
  v16bf a1 = load_a_tile(hb + (size_t)row0 * 64, 64, 32, lane);
  int c = lane & 15, hi = lane >> 4;
  for (int nt = 0; nt < 7; ++nt) {
    v8f acc = {};
    v16bf b0 = load_b_tile(wq + nt * 16 * 64, 64, 0, lane);
    v16bf b1 = load_b_tile(wq + nt * 16 * 64, 64, 32, lane);
    acc = WMMA_BF16(a0, b0, acc);
    acc = WMMA_BF16(a1, b1, acc);
    int col = nt * 16 + c;
    float bv = (col < 108) ? bias[col] : 0.f;
    for (int v = 0; v < 8; ++v) {
      int r = row0 + v + hi * 8;
      qkv[(size_t)r * 112 + col] = acc[v] + bv;
    }
  }
}

// ---- attention: one block per (b, head); Q/K/V staged in LDS ------------
__global__ void k_attn(const float* __restrict__ qkv, float* __restrict__ ao) {
  __shared__ float Qs[512 * 6], Ks[512 * 6], Vs[512 * 6];
  int b = blockIdx.x / 6, hd = blockIdx.x % 6;
  int tid = threadIdx.x;
  for (int i = tid; i < 512 * 6; i += 256) {
    int s = i / 6, d = i - 6 * s;
    size_t base = ((size_t)b * 512 + s) * 112;
    Qs[i] = qkv[base + hd * 6 + d];
    Ks[i] = qkv[base + 36 + hd * 6 + d];
    Vs[i] = qkv[base + 72 + hd * 6 + d];
  }
  __syncthreads();
  const float scale = 0.40824829046386302f;  // 1/sqrt(6)
  for (int q = tid; q < 512; q += 256) {
    float q0 = Qs[q * 6 + 0], q1 = Qs[q * 6 + 1], q2 = Qs[q * 6 + 2];
    float q3 = Qs[q * 6 + 3], q4 = Qs[q * 6 + 4], q5 = Qs[q * 6 + 5];
    float m = -3.4e38f, l = 0.f, o[6] = {0, 0, 0, 0, 0, 0};
    for (int k = 0; k < 512; ++k) {
      const float* kp = Ks + k * 6;
      float s = q0 * kp[0] + q1 * kp[1] + q2 * kp[2] + q3 * kp[3] + q4 * kp[4] + q5 * kp[5];
      s *= scale;
      float mn = fmaxf(m, s);
      float cf = __expf(m - mn);
      float p = __expf(s - mn);
      l = l * cf + p;
      const float* vp = Vs + k * 6;
      for (int d = 0; d < 6; ++d) o[d] = o[d] * cf + p * vp[d];
      m = mn;
    }
    float inv = 1.f / l;
    size_t row = (size_t)b * 512 + q;
    for (int d = 0; d < 6; ++d) ao[row * 36 + hd * 6 + d] = o[d] * inv;
  }
}

// ---- out_proj + residual + LN1 (VALU; 36x36 is tiny) --------------------
__global__ void k_oproj_ln(const float* __restrict__ ao, const float* __restrict__ ow,
                           const float* __restrict__ ob, const float* __restrict__ lw,
                           const float* __restrict__ lb, float* __restrict__ h,
                           unsigned short* __restrict__ hb) {
  __shared__ float red[256];
  int g = threadIdx.x >> 6, t = threadIdx.x & 63;
  int token = blockIdx.x * 4 + g;
  float val = 0.f;
  if (t < 36) {
    float acc = ob[t];
    const float* ap = ao + (size_t)token * 36;
    for (int j = 0; j < 36; ++j) acc += ap[j] * ow[t * 36 + j];
    val = h[(size_t)token * 36 + t] + acc;
  }
  red[threadIdx.x] = val;
  __syncthreads();
  for (int off = 32; off; off >>= 1) {
    if (t < off) red[threadIdx.x] += red[threadIdx.x + off];
    __syncthreads();
  }
  float mu = red[threadIdx.x & ~63] * (1.f / 36.f);
  __syncthreads();
  float dv = (t < 36) ? (val - mu) : 0.f;
  red[threadIdx.x] = dv * dv;
  __syncthreads();
  for (int off = 32; off; off >>= 1) {
    if (t < off) red[threadIdx.x] += red[threadIdx.x + off];
    __syncthreads();
  }
  float var = red[threadIdx.x & ~63] * (1.f / 36.f);
  if (t < 36) {
    float y = (val - mu) * rsqrtf(var + LN_EPS) * lw[t] + lb[t];
    h[(size_t)token * 36 + t] = y;
    hb[(size_t)token * 64 + t] = f32_to_bf16_bits(y);
  } else {
    hb[(size_t)token * 64 + t] = 0;
  }
}

// ---- residual + LN2 -----------------------------------------------------
__global__ void k_addln(const float* __restrict__ y2, const float* __restrict__ lw,
                        const float* __restrict__ lb, float* __restrict__ h,
                        unsigned short* __restrict__ hb) {
  __shared__ float red[256];
  int g = threadIdx.x >> 6, t = threadIdx.x & 63;
  int token = blockIdx.x * 4 + g;
  float val = 0.f;
  if (t < 36) val = h[(size_t)token * 36 + t] + y2[(size_t)token * 36 + t];
  red[threadIdx.x] = val;
  __syncthreads();
  for (int off = 32; off; off >>= 1) {
    if (t < off) red[threadIdx.x] += red[threadIdx.x + off];
    __syncthreads();
  }
  float mu = red[threadIdx.x & ~63] * (1.f / 36.f);
  __syncthreads();
  float dv = (t < 36) ? (val - mu) : 0.f;
  red[threadIdx.x] = dv * dv;
  __syncthreads();
  for (int off = 32; off; off >>= 1) {
    if (t < off) red[threadIdx.x] += red[threadIdx.x + off];
    __syncthreads();
  }
  float var = red[threadIdx.x & ~63] * (1.f / 36.f);
  if (t < 36) {
    float y = (val - mu) * rsqrtf(var + LN_EPS) * lw[t] + lb[t];
    h[(size_t)token * 36 + t] = y;
    hb[(size_t)token * 64 + t] = f32_to_bf16_bits(y);
  } else {
    hb[(size_t)token * 64 + t] = 0;
  }
}

// ---- fused FF: y2 = relu(X@W1^T+b1)@W2^T (+b2 in epilogue) --------------
// Block = 64 tokens, 8 waves. Streams d_ff=2048 in 16 chunks of 128 via LDS.
__global__ void k_ff(const unsigned short* __restrict__ hb,
                     const unsigned short* __restrict__ w1, const float* __restrict__ b1,
                     const unsigned short* __restrict__ w2, const float* __restrict__ b2,
                     float* __restrict__ y2) {
  __shared__ unsigned short Xs[64 * 64];    // 8 KB
  __shared__ unsigned short Hs[64 * 128];   // 16 KB
  int tid = threadIdx.x, wave = tid >> 5, lane = tid & 31;
  int c = lane & 15, hi = lane >> 4;
  int m0 = blockIdx.x * 64;
  // stage X block (bf16, padded K=64)
  for (int i = tid; i < 512; i += 256)
    ((uint4*)Xs)[i] = ((const uint4*)(hb + (size_t)m0 * 64))[i];
  __syncthreads();

  v8f accY[2] = {{}, {}};                   // tasks: wave, wave+8 (12 valid)
  for (int cn = 0; cn < 16; ++cn) {
    // phase A: H chunk = relu(X @ W1^T[:, cn*128 : cn*128+128]) -> Hs
    {
      int ncol0 = cn * 128 + wave * 16;
      v16bf b0 = load_b_tile(w1 + (size_t)ncol0 * 64, 64, 0, lane);
      v16bf b1v = load_b_tile(w1 + (size_t)ncol0 * 64, 64, 32, lane);
      float bv = b1[ncol0 + c];
      for (int mt = 0; mt < 4; ++mt) {
        v16bf a0 = load_a_tile(Xs + mt * 16 * 64, 64, 0, lane);
        v16bf a1 = load_a_tile(Xs + mt * 16 * 64, 64, 32, lane);
        v8f acc = {};
        acc = WMMA_BF16(a0, b0, acc);
        acc = WMMA_BF16(a1, b1v, acc);
        for (int v = 0; v < 8; ++v) {
          float r = fmaxf(acc[v] + bv, 0.f);
          int rr = mt * 16 + v + hi * 8;
          Hs[rr * 128 + wave * 16 + c] = f32_to_bf16_bits(r);
        }
      }
    }
    __syncthreads();
    // phase B: accumulate Y2 += Hchunk @ W2^T[chunk, :48]
    for (int ti = 0; ti < 2; ++ti) {
      int task = wave + ti * 8;
      if (task < 12) {
        int nt2 = task >> 2, mt = task & 3;
        for (int ks = 0; ks < 4; ++ks) {
          v16bf a = load_a_tile(Hs + mt * 16 * 128, 128, ks * 32, lane);
          v16bf b = load_b_tile(w2 + (size_t)(nt2 * 16) * 2048, 2048,
                                cn * 128 + ks * 32, lane);
          accY[ti] = WMMA_BF16(a, b, accY[ti]);
        }
      }
    }
    __syncthreads();
  }
  // epilogue: store valid 36 cols (bias added here)
  for (int ti = 0; ti < 2; ++ti) {
    int task = wave + ti * 8;
    if (task < 12) {
      int nt2 = task >> 2, mt = task & 3;
      int n2 = nt2 * 16 + c;
      if (n2 < 36) {
        float bv = b2[n2];
        for (int v = 0; v < 8; ++v) {
          int r = m0 + mt * 16 + v + hi * 8;
          y2[(size_t)r * 36 + n2] = accY[ti][v] + bv;
        }
      }
    }
  }
}

// ---- final mean over sequence ------------------------------------------
__global__ void k_mean(const float* __restrict__ h, float* __restrict__ out) {
  int b = blockIdx.x, d = threadIdx.x;
  if (d >= 36) return;
  float s = 0.f;
  for (int i = 0; i < 512; ++i) s += h[((size_t)b * 512 + i) * 36 + d];
  out[b * 36 + d] = s * (1.f / 512.f);
}

extern "C" void kernel_launch(void* const* d_in, const int* in_sizes, int n_in,
                              void* d_out, int out_size, void* d_ws, size_t ws_size,
                              hipStream_t stream) {
  (void)in_sizes; (void)n_in; (void)out_size; (void)ws_size;
  const int NTOK = 64 * 512;
  const float* x     = (const float*)d_in[0];
  const float* emb_w = (const float*)d_in[1];
  const float* emb_b = (const float*)d_in[2];

  char* w = (char*)d_ws;
  size_t off = 0;
  auto alloc = [&](size_t bytes) { char* p = w + off; off = (off + bytes + 255) & ~(size_t)255; return p; };
  unsigned short* hb   = (unsigned short*)alloc((size_t)NTOK * 64 * 2);
  float*          h    = (float*)alloc((size_t)NTOK * 36 * 4);
  float*          qkv  = (float*)alloc((size_t)NTOK * 112 * 4);
  float*          ao   = (float*)alloc((size_t)NTOK * 36 * 4);
  float*          y2   = (float*)alloc((size_t)NTOK * 36 * 4);
  unsigned short* wqp[2], *w1p[2], *w2p[2];
  for (int l = 0; l < 2; ++l) {
    wqp[l] = (unsigned short*)alloc(128 * 64 * 2);
    w1p[l] = (unsigned short*)alloc(2048 * 64 * 2);
    w2p[l] = (unsigned short*)alloc(48 * 2048 * 2);
  }

  // weight prep (pad + bf16 convert)
  for (int l = 0; l < 2; ++l) {
    const float* in_w  = (const float*)d_in[3 + 12 * l + 0];
    const float* ff1_w = (const float*)d_in[3 + 12 * l + 4];
    const float* ff2_w = (const float*)d_in[3 + 12 * l + 6];
    k_pad_bf16<<<(128 * 64 + 255) / 256, 256, 0, stream>>>(wqp[l], in_w, 108, 36, 128, 64);
    k_pad_bf16<<<(2048 * 64 + 255) / 256, 256, 0, stream>>>(w1p[l], ff1_w, 2048, 36, 2048, 64);
    k_pad_bf16<<<(48 * 2048 + 255) / 256, 256, 0, stream>>>(w2p[l], ff2_w, 36, 2048, 48, 2048);
  }

  k_embed<<<(NTOK * 64 + 255) / 256, 256, 0, stream>>>(x, emb_w, emb_b, h, hb);

  for (int l = 0; l < 2; ++l) {
    const float* in_b  = (const float*)d_in[3 + 12 * l + 1];
    const float* out_w = (const float*)d_in[3 + 12 * l + 2];
    const float* out_b = (const float*)d_in[3 + 12 * l + 3];
    const float* ff1_b = (const float*)d_in[3 + 12 * l + 5];
    const float* ff2_b = (const float*)d_in[3 + 12 * l + 7];
    const float* ln1_w = (const float*)d_in[3 + 12 * l + 8];
    const float* ln1_b = (const float*)d_in[3 + 12 * l + 9];
    const float* ln2_w = (const float*)d_in[3 + 12 * l + 10];
    const float* ln2_b = (const float*)d_in[3 + 12 * l + 11];

    k_qkv<<<NTOK / 128, 256, 0, stream>>>(hb, wqp[l], in_b, qkv);
    k_attn<<<64 * 6, 256, 0, stream>>>(qkv, ao);
    k_oproj_ln<<<NTOK / 4, 256, 0, stream>>>(ao, out_w, out_b, ln1_w, ln1_b, h, hb);
    k_ff<<<NTOK / 64, 256, 0, stream>>>(hb, w1p[l], ff1_b, w2p[l], ff2_b, y2);
    k_addln<<<NTOK / 4, 256, 0, stream>>>(y2, ln2_w, ln2_b, h, hb);
  }

  k_mean<<<64, 64, 0, stream>>>(h, (float*)d_out);
}
